// RnnBlock_20040317403800
// MI455X (gfx1250) — compile-verified, matrix operand order checked by reading, not run on hardware
//
#include <hip/hip_runtime.h>
#include <hip/hip_bf16.h>
#include <stdint.h>

// Problem constants (match reference)
#define B_ 16
#define T_ 1600
#define D_ 512
#define U_ 1024
// K total = D + U = 1536 -> 48 k-blocks of 32
#define KBLK 48
#define KXBLK 16              // first 16 blocks come from x, last 32 from h
#define NWG_DIR 64            // workgroups per direction (each owns 16 units)

typedef __attribute__((ext_vector_type(16))) __bf16 v16bf;
typedef __attribute__((ext_vector_type(8)))  float  v8f;

// Workspace byte offsets
#define WP_OFF   0ul           // packed weights: 2 dir * 64 utile * 4 gate * 48 blk * 512 bf16 = 25,165,824 B
#define AXP_OFF  25165824ul    // packed x A-fragments: 1600 * 16 blk * 512 bf16 = 26,214,400 B
#define AHP_OFF  51380224ul    // packed h A-fragments: 2 dir * 32 blk * 512 bf16 = 65,536 B
#define SYNC_OFF 51445760ul    // barrier counters

__device__ __forceinline__ unsigned short f2bf(float f) {
  unsigned u = __float_as_uint(f);
  u += 0x7FFFu + ((u >> 16) & 1u);         // round-to-nearest-even
  return (unsigned short)(u >> 16);
}
__device__ __forceinline__ float sigm(float x) { return 1.0f / (1.0f + __expf(-x)); }

__device__ __forceinline__ v16bf ldfrag(const unsigned short* p) {
  return *(const v16bf*)p;
}

// ---------------------------------------------------------------------------
// Pack combined weights Wc[k,n] (k<512: kernel, k>=512: rkernel) into the
// WMMA 16-bit B-operand fragment layout:
//   frag[block][lane*16 + e] = Wc[block*32 + (e>>3)*16 + (lane>>4)*8 + (e&7),
//                                 gate*1024 + utile*16 + (lane&15)]
// Linear dest: [dir][utile(64)][gate(4)][block(48)][lane(32)][e(16)]
// ---------------------------------------------------------------------------
__global__ void pack_weights_kernel(const float* __restrict__ kf, const float* __restrict__ rf,
                                    const float* __restrict__ kb, const float* __restrict__ rb,
                                    unsigned short* __restrict__ Wp) {
  const size_t PER_DIR = (size_t)NWG_DIR * 4 * KBLK * 512;   // 6,291,456
  size_t idx = (size_t)blockIdx.x * 256 + threadIdx.x;
  if (idx >= 2 * PER_DIR) return;
  int d = idx >= PER_DIR;
  size_t r = idx - (size_t)d * PER_DIR;
  int e = (int)(r & 15);
  int l = (int)((r >> 4) & 31);
  size_t blkLin = r >> 9;                  // (utile*4 + gate)*48 + block
  int block = (int)(blkLin % KBLK);
  size_t jg = blkLin / KBLK;
  int g = (int)(jg & 3);
  int j = (int)(jg >> 2);
  int n = g * 1024 + j * 16 + (l & 15);
  int K = block * 32 + ((e >> 3) << 4) + ((l >> 4) << 3) + (e & 7);
  const float* kk = d ? kb : kf;
  const float* rr = d ? rb : rf;
  float v = (K < D_) ? kk[(size_t)K * (4 * U_) + n]
                     : rr[(size_t)(K - D_) * (4 * U_) + n];
  Wp[idx] = f2bf(v);
}

// ---------------------------------------------------------------------------
// Pack x into 16-bit A-operand fragments, one set of 16 k-blocks per timestep:
//   Axp[t*8192 + block*512 + lane*16 + e] =
//       x[m = lane&15, t, K = block*32 + (e>>3)*16 + (lane>>4)*8 + (e&7)]
// Shared by both directions (backward just reads t' = T-1-s).
// ---------------------------------------------------------------------------
__global__ void pack_x_kernel(const float* __restrict__ x, unsigned short* __restrict__ Axp) {
  size_t idx = (size_t)blockIdx.x * 256 + threadIdx.x;
  if (idx >= (size_t)T_ * KXBLK * 512) return;
  int e = (int)(idx & 15);
  int l = (int)((idx >> 4) & 31);
  int block = (int)((idx >> 9) & 15);
  int t = (int)(idx >> 13);
  int m = l & 15;
  int K = block * 32 + ((e >> 3) << 4) + ((l >> 4) << 3) + (e & 7);
  Axp[idx] = f2bf(x[((size_t)m * T_ + t) * D_ + K]);
}

// ---------------------------------------------------------------------------
// Zero packed-h (h0 = 0) and barrier counters every launch (graph-replay safe);
// also emit the x_len tail of the tuple output.
// ---------------------------------------------------------------------------
__global__ void init_state_kernel(unsigned short* __restrict__ Ahp, unsigned* __restrict__ sync,
                                  const int* __restrict__ x_len, float* __restrict__ out) {
  for (int i = threadIdx.x; i < 2 * 32 * 512; i += 256) Ahp[i] = 0;
  if (threadIdx.x < 16) sync[threadIdx.x] = 0u;
  if (threadIdx.x < B_)
    ((int*)out)[(size_t)B_ * T_ * 2 * U_ + threadIdx.x] = x_len[threadIdx.x];
}

// ---------------------------------------------------------------------------
// Persistent bidirectional LSTM scan.
// grid = 128 blocks (dir = blockIdx.x>>6, unit-tile j = blockIdx.x&63),
// 256 threads = 8 waves: wave w -> gate g = w>>1, k-half kh = w&1 (24 blocks).
// Weights live in LDS for the whole scan (loaded via async-to-LDS).
// ---------------------------------------------------------------------------
__global__ void lstm_scan_kernel(const unsigned short* __restrict__ Wp,
                                 const unsigned short* __restrict__ Axp,
                                 unsigned short* __restrict__ Ahp,
                                 const float* __restrict__ bias_f,
                                 const float* __restrict__ bias_b,
                                 float* __restrict__ out,
                                 unsigned* __restrict__ sync) {
  extern __shared__ __align__(16) char smem[];
  unsigned short* Wlds   = (unsigned short*)smem;                // 4*48*512 bf16 = 196608 B
  float*          gates  = (float*)(smem + 196608);              // 8 partial 16x16 tiles = 8192 B
  float*          cstate = (float*)(smem + 196608 + 8192);       // 16x16 f32 = 1024 B

  const int tid  = threadIdx.x;
  const int lane = tid & 31;
  const int dir  = blockIdx.x >> 6;
  const int j    = blockIdx.x & (NWG_DIR - 1);
  // wave-uniform role values, forced scalar so loop bounds/branches are SALU
  const int wS  = __builtin_amdgcn_readfirstlane(tid >> 5);
  const int gS  = wS >> 1;                 // gate handled by this wave
  const int khS = wS & 1;                  // k-half: 0 -> blocks 0..23, 1 -> 24..47
  const int u0  = j * 16;

  // ---- one-time: async-copy this WG's 192KB weight slice into LDS ----
  const unsigned short* wsrc = Wp + (size_t)dir * ((size_t)NWG_DIR * 4 * KBLK * 512)
                                  + (size_t)j * (4 * KBLK * 512);
  {
    unsigned ldsbase = (unsigned)(uintptr_t)Wlds;                // low 32 bits = LDS byte offset
    unsigned long long gbase = (unsigned long long)(uintptr_t)wsrc;
    for (int i = tid; i < (4 * KBLK * 512 * 2) / 16; i += 256) { // 12288 x 16B
      unsigned ld = ldsbase + (unsigned)i * 16u;
      unsigned long long ga = gbase + (unsigned long long)i * 16ull;
      asm volatile("global_load_async_to_lds_b128 %0, %1, off" :: "v"(ld), "v"(ga) : "memory");
    }
#if __has_builtin(__builtin_amdgcn_s_wait_asynccnt)
    __builtin_amdgcn_s_wait_asynccnt(0);
#else
    asm volatile("s_wait_asynccnt 0" ::: "memory");
#endif
  }
  cstate[tid] = 0.0f;
  __syncthreads();

  const float* bias = dir ? bias_b : bias_f;
  const float breg = (khS == 0) ? bias[gS * 1024 + u0 + (lane & 15)] : 0.0f;

  unsigned* count = sync + dir * 8;                   // per-direction cumulative barrier
  unsigned short* AhD = Ahp + dir * (32 * 512);
  const int mhi  = (lane >> 4) * 8;                   // C/D tile: VGPR v -> row M = v + mhi
  const int ncol = lane & 15;

  // loop partition (scalar): kh=0 -> 16 x-blocks then 8 h-blocks; kh=1 -> 24 h-blocks
  const int NX = khS ? 0 : KXBLK;                     // x-fragment count
  const int NH = 24 - NX;                             // h-fragment count
  const unsigned short* bbase = Wlds + ((size_t)(gS * KBLK + khS * 24) * 512 + lane * 16);
  const unsigned short* ahbase = AhD + (size_t)(khS ? 8 : 0) * 512 + lane * 16;

  for (int s = 0; s < T_; ++s) {
    const int tx = dir ? (T_ - 1 - s) : s;            // time index into x and out

    // prefetch next step's x fragments while we compute
    if (s + 1 < T_) {
      int txn = dir ? (T_ - 2 - s) : (s + 1);
      __builtin_prefetch(Axp + (size_t)txn * (KXBLK * 512) + (size_t)(khS * 8) * 512 + lane * 16, 0, 1);
    }

    // ---- WMMA: this wave's 24 k-blocks of gate g (double-buffered loads) ----
    v8f acc = {breg, breg, breg, breg, breg, breg, breg, breg};
    const unsigned short* ap = (NX > 0)
        ? (Axp + (size_t)tx * (KXBLK * 512) + lane * 16)   // start in x fragments
        : ahbase;                                          // or straight into h fragments
    const unsigned short* bp = bbase;

    v16bf a = ldfrag(ap);
    v16bf b = ldfrag(bp);
#pragma unroll 4
    for (int i = 1; i < 24; ++i) {
      // advance A pointer; switch from x-region to h-region at i == NX
      ap = (i == NX) ? (ahbase) : (ap + 512);
      bp += 512;
      v16bf an = ldfrag(ap);                // next fragments in flight...
      v16bf bn = ldfrag(bp);
      acc = __builtin_amdgcn_wmma_f32_16x16x32_bf16(false, a, false, b, (short)0, acc,
                                                    false, false);   // ...while this WMMA runs
      a = an; b = bn;
    }
    acc = __builtin_amdgcn_wmma_f32_16x16x32_bf16(false, a, false, b, (short)0, acc,
                                                  false, false);

    // ---- stash partial gate tile [M=batch][N=unit] in LDS ----
    {
      float* gw = gates + wS * 256;
#pragma unroll
      for (int v = 0; v < 8; ++v) gw[(v + mhi) * 16 + ncol] = acc[v];
    }
    __syncthreads();

    // ---- elementwise LSTM cell update: one (batch, unit) per thread ----
    {
      int b_ = tid >> 4, ul = tid & 15;
      int idx = b_ * 16 + ul;
      float i_  = gates[0 * 256 + idx] + gates[1 * 256 + idx];
      float f_  = gates[2 * 256 + idx] + gates[3 * 256 + idx];
      float cc_ = gates[4 * 256 + idx] + gates[5 * 256 + idx];
      float o_  = gates[6 * 256 + idx] + gates[7 * 256 + idx];
      float c = cstate[idx];
      c = sigm(f_) * c + sigm(i_) * tanhf(cc_);
      float h = sigm(o_) * tanhf(c);
      cstate[idx] = c;
      // fp32 output [B, T, 2U], backward half at column offset U
      out[((size_t)b_ * T_ + tx) * (2 * U_) + dir * U_ + u0 + ul] = h;
      // bf16 h written directly in packed A-fragment layout for next step
      int u = u0 + ul;
      int blk = u >> 5, kl = u & 31;
      int e = ((kl >> 4) << 3) + (kl & 7);
      int lL = (((kl >> 3) & 1) << 4) + b_;
      AhD[(size_t)blk * 512 + lL * 16 + e] = f2bf(h);
    }
    __syncthreads();

    // ---- per-direction grid barrier (cumulative counter, zeroed per launch) ----
    __threadfence();
    if (tid == 0) {
      unsigned target = (unsigned)(s + 1) * (unsigned)NWG_DIR;
      __hip_atomic_fetch_add(count, 1u, __ATOMIC_ACQ_REL, __HIP_MEMORY_SCOPE_AGENT);
      while (__hip_atomic_load(count, __ATOMIC_ACQUIRE, __HIP_MEMORY_SCOPE_AGENT) < target)
        __builtin_amdgcn_s_sleep(1);
    }
    __syncthreads();
  }
}

// ---------------------------------------------------------------------------
extern "C" void kernel_launch(void* const* d_in, const int* in_sizes, int n_in,
                              void* d_out, int out_size, void* d_ws, size_t ws_size,
                              hipStream_t stream) {
  (void)in_sizes; (void)n_in; (void)out_size; (void)ws_size;
  const float* x     = (const float*)d_in[0];
  const int*   x_len = (const int*)d_in[1];
  const float* kf    = (const float*)d_in[2];
  const float* rf    = (const float*)d_in[3];
  const float* bf_   = (const float*)d_in[4];
  const float* kb    = (const float*)d_in[5];
  const float* rb    = (const float*)d_in[6];
  const float* bb_   = (const float*)d_in[7];
  float* out = (float*)d_out;
  char*  ws  = (char*)d_ws;

  unsigned short* Wp   = (unsigned short*)(ws + WP_OFF);
  unsigned short* Axp  = (unsigned short*)(ws + AXP_OFF);
  unsigned short* Ahp  = (unsigned short*)(ws + AHP_OFF);
  unsigned*       sync = (unsigned*)(ws + SYNC_OFF);

  {
    size_t n = 2ul * NWG_DIR * 4 * KBLK * 512;
    pack_weights_kernel<<<(unsigned)((n + 255) / 256), 256, 0, stream>>>(kf, rf, kb, rb, Wp);
  }
  {
    size_t n = (size_t)T_ * KXBLK * 512;
    pack_x_kernel<<<(unsigned)((n + 255) / 256), 256, 0, stream>>>(x, Axp);
  }
  init_state_kernel<<<1, 256, 0, stream>>>(Ahp, sync, x_len, out);

  const size_t smem = 196608 + 8192 + 1024;   // weights + gate partials + cell state
  lstm_scan_kernel<<<2 * NWG_DIR, 256, smem, stream>>>(Wp, Axp, Ahp, bf_, bb_, out, sync);
}
